// VQ_VAE_54855322304643
// MI455X (gfx1250) — matrix operand order repaired and checked by reading, not compiled, not run on hardware
//
#include <hip/hip_runtime.h>

// ---------------------------------------------------------------------------
// CDNA5 (gfx1250) VQ-VAE forward: implicit-GEMM convs + VQ codebook GEMM.
// Matrix math via v_wmma_f32_16x16x32_f16 (wave32); weight tiles staged into
// LDS by the Tensor Data Mover (tensor_load_to_lds + s_wait_tensorcnt).
// ---------------------------------------------------------------------------

typedef __attribute__((ext_vector_type(16))) _Float16 v16h;
typedef __attribute__((ext_vector_type(8)))  _Float16 v8h;
typedef __attribute__((ext_vector_type(8)))  float    v8f;
typedef unsigned int u32;
typedef __attribute__((ext_vector_type(4))) u32 v4u;
typedef __attribute__((ext_vector_type(4))) int v4i;
typedef __attribute__((ext_vector_type(8))) int v8i;

#if __has_builtin(__builtin_amdgcn_tensor_load_to_lds)
#define HAVE_TDM 1
#else
#define HAVE_TDM 0
#endif

// ---- WMMA fragment loaders (ISA 7.12.2 layouts, wave32) -------------------
// A (16x32 f16): lanes 0-15 = rows M0..15 with K {kb..kb+7, 16+kb..16+kb+7},
// kb = 0 for lanes<16, kb = 8 for lanes>=16.
__device__ inline v16h load_frag_a(const _Float16* base, int ldk) {
  const int lane = threadIdx.x & 31;
  const int row  = lane & 15;
  const int kb   = (lane >> 4) ? 8 : 0;
  const _Float16* p = base + row * ldk;
  v8h lo = *(const v8h*)(p + kb);        // K = kb .. kb+7
  v8h hi = *(const v8h*)(p + 16 + kb);   // K = 16+kb .. 16+kb+7
  v16h r;
#pragma unroll
  for (int i = 0; i < 8; ++i) { r[i] = lo[i]; r[8 + i] = hi[i]; }
  return r;
}

// B (32x16 f16), tile stored [n][k]: lane holds column n=lane%16, K
// {kb..kb+7, 16+kb..16+kb+7}, kb = (lane<16) ? 0 : 8.
__device__ inline v16h load_frag_b(const _Float16* base, int ldk) {
  const int lane = threadIdx.x & 31;
  const int col  = lane & 15;
  const int kb   = (lane >> 4) ? 8 : 0;
  const _Float16* p = base + col * ldk + kb;
  v8h lo = *(const v8h*)(p);
  v8h hi = *(const v8h*)(p + 16);
  v16h r;
#pragma unroll
  for (int i = 0; i < 8; ++i) { r[i] = lo[i]; r[8 + i] = hi[i]; }
  return r;
}

// ---- TDM: async-load a [rem_n x rem_k]-clamped 64x32 f16 tile to LDS ------
// D# per ISA ch.8: 2-D tensor, data_size=2B, tile_dim0=32 (k, contiguous),
// tile_dim1=64 (rows), row stride = ldk elements. pad_enable with
// pad_interval=16 DWORDs (one 64B row) and pad_amount=4 DWORDs reproduces the
// 40-half (80B) padded LDS row stride. OOB rows/cols return zero.
__device__ inline void tdm_load_b_tile(const _Float16* gsrc, unsigned lds_off,
                                       int rem_k, int rem_n, int ldk) {
#if HAVE_TDM
  const unsigned long long ga = (unsigned long long)(uintptr_t)gsrc;
  v4u g0;
  g0[0] = 1u;                                            // count=1 (valid user D#)
  g0[1] = lds_off;                                       // lds_addr (bytes)
  g0[2] = (u32)(ga & 0xffffffffu);                       // global_addr[31:0]
  g0[3] = (u32)((ga >> 32) & 0x1ffffffu) | (2u << 30);   // addr[56:32] | type=2
  const u32 d0 = (u32)rem_k;                             // tensor_dim0
  const u32 d1 = (u32)rem_n;                             // tensor_dim1
  const u32 st = (u32)ldk;                               // tensor_dim0_stride
  v8i g1;
  g1[0] = (int)((1u << 16) | (1u << 20) | (3u << 22) | (3u << 25)); // 2B, pad en, 16DW/4DW
  g1[1] = (int)((d0 & 0xffffu) << 16);                   // [47:32]=0, tensor_dim0 lo
  g1[2] = (int)((d0 >> 16) | ((d1 & 0xffffu) << 16));    // dim0 hi, dim1 lo
  g1[3] = (int)((d1 >> 16) | (32u << 16));               // dim1 hi, tile_dim0=32
  g1[4] = (int)64u;                                      // tile_dim1=64, tile_dim2=0
  g1[5] = (int)st;                                       // dim0_stride[31:0]
  g1[6] = 0;                                             // dim0_stride hi, dim1_stride lo
  g1[7] = 0;                                             // dim1_stride hi
  const v4i z4 = {0, 0, 0, 0};
#if __clang_major__ >= 23
  const v8i z8 = {0, 0, 0, 0, 0, 0, 0, 0};
  __builtin_amdgcn_tensor_load_to_lds(g0, g1, z4, z4, z8, 0);
#else
  __builtin_amdgcn_tensor_load_to_lds(g0, g1, z4, z4, 0);
#endif
  __builtin_amdgcn_s_wait_tensorcnt(0);
#else
  (void)gsrc; (void)lds_off; (void)rem_k; (void)rem_n; (void)ldk;
#endif
}

// ---------------------------------------------------------------------------
// Generic implicit-GEMM convolution, templated on the static conv geometry so
// im2col index math lowers to mul/shift instead of runtime idiv.
//   M = Nb*Ho*Wo, N = Cout, K = Cin*KH*KW.
//   Block: 128 threads (4 waves), 64x64 tile, each wave 32x32 (2x2 WMMA).
// ---------------------------------------------------------------------------
template <int KH, int KW, int STRIDE, int PAD, int DIL>
__global__ __launch_bounds__(128) void conv_wmma_kernel(
    const float* __restrict__ in, const _Float16* __restrict__ w,
    const float* __restrict__ bias, const float* __restrict__ resid,
    float* __restrict__ out,
    int Nb, int Cin, int Hin, int Win,
    int Cout, int Ho, int Wo,
    int relu_in, int relu_out)
{
  constexpr int KHW = KH * KW;
  const int Ktot = Cin * KHW;
  const int m0 = blockIdx.x * 64;
  const int n0 = blockIdx.y * 64;

  __shared__ alignas(16) _Float16 aT[64 * 40];
  __shared__ alignas(16) _Float16 bT[64 * 40];

  const int tid  = threadIdx.x;
  const int lane = tid & 31;
  const int wv   = tid >> 5;
  const int wr   = (wv >> 1) * 32;
  const int wc   = (wv & 1) * 32;

  const v8f zero8 = {0.f, 0.f, 0.f, 0.f, 0.f, 0.f, 0.f, 0.f};
  v8f acc[2][2];
  acc[0][0] = zero8; acc[0][1] = zero8; acc[1][0] = zero8; acc[1][1] = zero8;

  const int arow  = tid >> 1;            // 0..63
  const int akoff = (tid & 1) * 16;
  const int m = m0 + arow;
  int nb, oy, ox;
  { const int hw = Ho * Wo; nb = m / hw; const int r = m % hw; oy = r / Wo; ox = r - (r / Wo) * Wo; }
  const int brow  = tid >> 1;
  const int bkoff = (tid & 1) * 16;
  const unsigned bT_lds = (unsigned)(uintptr_t)&bT[0];

  for (int k0 = 0; k0 < Ktot; k0 += 32) {
    // ---- stage A tile (im2col gather, fp32 -> fp16, optional relu) ----
#pragma unroll 4
    for (int j = 0; j < 16; ++j) {
      const int k = k0 + akoff + j;
      _Float16 v = (_Float16)0.f;
      if (k < Ktot) {
        const int c  = k / KHW;            // constant divisor -> mul/shift
        const int rr = k - c * KHW;
        const int ky = rr / KW, kx = rr - (rr / KW) * KW;
        const int vy = oy * STRIDE + ky - PAD;
        const int vx = ox * STRIDE + kx - PAD;
        bool ok = true; int iy = vy, ix = vx;
        if (DIL == 2) { if ((vy | vx) & 1) ok = false; iy = vy >> 1; ix = vx >> 1; }
        if (ok && iy >= 0 && iy < Hin && ix >= 0 && ix < Win) {
          float f = in[(((long)nb * Cin + c) * Hin + iy) * Win + ix];
          if (relu_in) f = fmaxf(f, 0.f);
          v = (_Float16)f;
        }
      }
      aT[arow * 40 + akoff + j] = v;
    }
    // ---- stage B tile: Tensor Data Mover (wave 0 issues + waits) ----
#if HAVE_TDM
    if (wv == 0) {
      tdm_load_b_tile(w + (long)n0 * Ktot + k0, bT_lds, Ktot - k0, Cout - n0, Ktot);
    }
#else
    {
      const int n = n0 + brow;
#pragma unroll 4
      for (int j = 0; j < 16; ++j) {
        const int k = k0 + bkoff + j;
        _Float16 v = (_Float16)0.f;
        if (n < Cout && k < Ktot) v = w[(long)n * Ktot + k];
        bT[brow * 40 + bkoff + j] = v;
      }
    }
#endif
    __syncthreads();

    v16h af[2], bf[2];
#pragma unroll
    for (int sm = 0; sm < 2; ++sm) af[sm] = load_frag_a(&aT[(wr + sm * 16) * 40], 40);
#pragma unroll
    for (int sn = 0; sn < 2; ++sn) bf[sn] = load_frag_b(&bT[(wc + sn * 16) * 40], 40);
#pragma unroll
    for (int sm = 0; sm < 2; ++sm)
#pragma unroll
      for (int sn = 0; sn < 2; ++sn)
        acc[sm][sn] = __builtin_amdgcn_wmma_f32_16x16x32_f16(
            false, af[sm], false, bf[sn], (short)0, acc[sm][sn], false, false);
    __syncthreads();
  }

  // ---- epilogue: bias + residual + relu, scatter to NCHW ----
  const int colLane = lane & 15;
  const int rowHi   = (lane >> 4) * 8;
#pragma unroll
  for (int sm = 0; sm < 2; ++sm)
#pragma unroll
    for (int sn = 0; sn < 2; ++sn) {
      const int cout = n0 + wc + sn * 16 + colLane;
      if (cout >= Cout) continue;
      const float bv = bias ? bias[cout] : 0.f;
#pragma unroll
      for (int r = 0; r < 8; ++r) {
        const int rowm = m0 + wr + sm * 16 + r + rowHi;
        float v = acc[sm][sn][r] + bv;
        int nb2, oy2, ox2;
        { const int hw = Ho * Wo; nb2 = rowm / hw; const int rr2 = rowm % hw;
          oy2 = rr2 / Wo; ox2 = rr2 - oy2 * Wo; }
        const long off = (((long)nb2 * Cout + cout) * Ho + oy2) * Wo + ox2;
        if (resid) v += resid[off];
        if (relu_out) v = fmaxf(v, 0.f);
        out[off] = v;
      }
    }
}

// ---------------------------------------------------------------------------
// VQ: argmin_k ||E_k||^2 - 2 x.E_k via WMMA (two 16x16x32 k-steps).
// Block = 64 threads (2 waves), each wave scans 256 codes.
// ---------------------------------------------------------------------------
__global__ __launch_bounds__(64) void vq_argmin_kernel(
    const float* __restrict__ Ze, const _Float16* __restrict__ Ef16,
    const float* __restrict__ Enorm, int* __restrict__ idx,
    int Nb, int D, int H, int W, int Kcodes)
{
  __shared__ alignas(16) _Float16 aT[16 * 72];
  __shared__ float redV[2][16];
  __shared__ int   redI[2][16];

  const int tid  = threadIdx.x;
  const int lane = tid & 31;
  const int wv   = tid >> 5;
  const int m0   = blockIdx.x * 16;

  { // stage 16 x 64 A tile (NCHW gather -> NHWC rows), fp32 -> fp16
    const int row = tid >> 2, koff = (tid & 3) * 16;
    const int mm = m0 + row;
    const int hw = H * W; const int nb = mm / hw; const int r = mm % hw;
    const int h = r / W; const int w_ = r - h * W;
#pragma unroll 4
    for (int j = 0; j < 16; ++j) {
      const int d = koff + j;
      aT[row * 72 + d] = (_Float16)Ze[(((long)nb * D + d) * H + h) * W + w_];
    }
  }
  __syncthreads();

  float bestv[8]; int besti[8];
#pragma unroll
  for (int r = 0; r < 8; ++r) { bestv[r] = 3.4e38f; besti[r] = 0x7fffffff; }
  const int half = lane >> 4, col = lane & 15;

  for (int t = 0; t < Kcodes / 32; ++t) {
    const int n0 = wv * (Kcodes / 2) + t * 16;
    v8f acc = {0.f, 0.f, 0.f, 0.f, 0.f, 0.f, 0.f, 0.f};
#pragma unroll
    for (int ks = 0; ks < 2; ++ks) {
      v16h af = load_frag_a(aT + ks * 32, 72);
      const _Float16* p = Ef16 + (long)(n0 + col) * D + ks * 32 + half * 8;
      v8h lo = *(const v8h*)p;
      v8h hi = *(const v8h*)(p + 16);
      v16h bf;
#pragma unroll
      for (int i = 0; i < 8; ++i) { bf[i] = lo[i]; bf[8 + i] = hi[i]; }
      acc = __builtin_amdgcn_wmma_f32_16x16x32_f16(
          false, af, false, bf, (short)0, acc, false, false);
    }
    const int n = n0 + col;
    const float en = Enorm[n];
#pragma unroll
    for (int r = 0; r < 8; ++r) {
      const float s = en - 2.0f * acc[r];
      if (s < bestv[r] || (s == bestv[r] && n < besti[r])) { bestv[r] = s; besti[r] = n; }
    }
  }

#pragma unroll
  for (int r = 0; r < 8; ++r) {
    float v = bestv[r]; int bi = besti[r];
    for (int off = 1; off < 16; off <<= 1) {
      const float ov = __shfl_xor(v, off, 32);
      const int   oi = __shfl_xor(bi, off, 32);
      if (ov < v || (ov == v && oi < bi)) { v = ov; bi = oi; }
    }
    if (col == 0) { redV[wv][half * 8 + r] = v; redI[wv][half * 8 + r] = bi; }
  }
  __syncthreads();
  if (tid < 16) {
    const float v0 = redV[0][tid]; const int i0 = redI[0][tid];
    const float v1 = redV[1][tid]; const int i1 = redI[1][tid];
    idx[m0 + tid] = (v1 < v0 || (v1 == v0 && i1 < i0)) ? i1 : i0;
  }
}

// ---------------------------------------------------------------------------
// Small helper kernels
// ---------------------------------------------------------------------------
__global__ void zero_kernel(float* p, int n) {
  const int i = blockIdx.x * blockDim.x + threadIdx.x;
  if (i < n) p[i] = 0.f;
}

__global__ void f32_to_f16_kernel(const float* __restrict__ s, _Float16* __restrict__ d, int n) {
  const int i = blockIdx.x * blockDim.x + threadIdx.x;
  if (i < n) d[i] = (_Float16)s[i];
}

__global__ void enorm_kernel(const float* __restrict__ E, float* __restrict__ Enorm, int K, int D) {
  const int k = blockIdx.x * blockDim.x + threadIdx.x;
  if (k < K) {
    float s = 0.f;
    for (int d = 0; d < D; ++d) { const float v = E[(long)k * D + d]; s += v * v; }
    Enorm[k] = s;
  }
}

// ConvTranspose weights (Cin,Cout,k,k) -> flipped f16 (Cout,Cin,k,k)
__global__ void wtrans_kernel(const float* __restrict__ w, _Float16* __restrict__ out,
                              int Cin, int Cout, int Kk) {
  const int i = blockIdx.x * blockDim.x + threadIdx.x;
  const int total = Cin * Cout * Kk * Kk;
  if (i >= total) return;
  const int kx = i % Kk; int t = i / Kk;
  const int ky = t % Kk; t /= Kk;
  const int co = t % Cout; const int ci = t / Cout;
  out[(((long)co * Cin + ci) * Kk + (Kk - 1 - ky)) * Kk + (Kk - 1 - kx)] = (_Float16)w[i];
}

// Gather Zq (NCHW) from codebook + accumulate sum((Zq-Zt)^2)
__global__ void vq_gather_loss_kernel(const float* __restrict__ Ze, const float* __restrict__ E,
                                      const int* __restrict__ idx, float* __restrict__ Zq,
                                      float* __restrict__ accum, int total, int D, int H, int W) {
  __shared__ float sred[256];
  float local = 0.f;
  for (int i = blockIdx.x * blockDim.x + threadIdx.x; i < total; i += gridDim.x * blockDim.x) {
    const int w_ = i % W; int t1 = i / W;
    const int h = t1 % H; int t2 = t1 / H;
    const int d = t2 % D; const int nb = t2 / D;
    const int m = (nb * H + h) * W + w_;
    const float e = E[(long)idx[m] * D + d];
    const float z = Ze[i];
    Zq[i] = e;
    const float df = e - z;
    local += df * df;
  }
  sred[threadIdx.x] = local;
  __syncthreads();
  for (int s = 128; s > 0; s >>= 1) {
    if ((int)threadIdx.x < s) sred[threadIdx.x] += sred[threadIdx.x + s];
    __syncthreads();
  }
  if (threadIdx.x == 0) atomicAdd(accum, sred[0]);
}

__global__ void hist_kernel(const int* __restrict__ idx, float* __restrict__ counts, int n) {
  const int i = blockIdx.x * blockDim.x + threadIdx.x;
  if (i < n) atomicAdd(&counts[idx[i]], 1.0f);
}

__global__ void finalize_kernel(const float* __restrict__ counts, const float* __restrict__ accum,
                                float* __restrict__ out, long R, int K, float total, float denom) {
  __shared__ float sred[512];
  const int t = threadIdx.x;
  float term = 0.f;
  if (t < K) {
    const float p = counts[t] / total;
    term = -p * log2f(p + 1e-10f);
  }
  sred[t] = term;
  __syncthreads();
  for (int s = 256; s > 0; s >>= 1) {
    if (t < s) sred[t] += sred[t + s];
    __syncthreads();
  }
  if (t == 0) {
    const float Hb  = sred[0];
    const float mse = accum[0] / denom;
    out[0]         = 1.25f * mse;  // q + BETA*e, numerically e==q
    out[R + 1]     = mse;          // e_latent_loss
    out[R + 2]     = mse;          // q_latent_loss
    out[R + 3]     = exp2f(Hb);    // est_words
  }
}

// ---------------------------------------------------------------------------
// Host orchestration
// ---------------------------------------------------------------------------
template <int KH, int KW, int S, int P, int D>
static inline void conv(hipStream_t stream, const float* in, const _Float16* w,
                        const float* b, const float* res, float* o,
                        int Nb, int Cin, int Hin, int Win, int Cout, int Ho, int Wo,
                        int ri, int ro) {
  dim3 g((unsigned)((Nb * Ho * Wo) / 64), (unsigned)((Cout + 63) / 64));
  conv_wmma_kernel<KH, KW, S, P, D><<<g, 128, 0, stream>>>(
      in, w, b, res, o, Nb, Cin, Hin, Win, Cout, Ho, Wo, ri, ro);
}

extern "C" void kernel_launch(void* const* d_in, const int* in_sizes, int n_in,
                              void* d_out, int out_size, void* d_ws, size_t ws_size,
                              hipStream_t stream) {
  (void)in_sizes; (void)n_in; (void)out_size; (void)ws_size;
  const float* x        = (const float*)d_in[0];
  const float* enc_w1   = (const float*)d_in[1];
  const float* enc_b1   = (const float*)d_in[2];
  const float* enc_w2   = (const float*)d_in[3];
  const float* enc_b2   = (const float*)d_in[4];
  const float* enc_w3   = (const float*)d_in[5];
  const float* enc_b3   = (const float*)d_in[6];
  const float* enc_w4   = (const float*)d_in[7];
  const float* enc_b4   = (const float*)d_in[8];
  const float* enc_res_w1 = (const float*)d_in[9];
  const float* enc_res_w2 = (const float*)d_in[10];
  const float* enc_adj_w  = (const float*)d_in[11];
  const float* enc_adj_b  = (const float*)d_in[12];
  const float* E          = (const float*)d_in[13];
  const float* dec_adj_w  = (const float*)d_in[14];
  const float* dec_adj_b  = (const float*)d_in[15];
  const float* dec_res_w1 = (const float*)d_in[16];
  const float* dec_res_w2 = (const float*)d_in[17];
  const float* tc1_w      = (const float*)d_in[18];
  const float* tc1_b      = (const float*)d_in[19];
  const float* tc2_w      = (const float*)d_in[20];
  const float* tc2_b      = (const float*)d_in[21];
  float* out = (float*)d_out;

  const int Bn = 64, Dd = 64, Kc = 512;
  const long R = 64L * 3 * 128 * 128;  // 3145728 recon elements

  // workspace carve-up
  size_t off = 0;
  auto alloc = [&](size_t bytes) -> void* {
    void* p = (char*)d_ws + off;
    off += (bytes + 255) & ~(size_t)255;
    return p;
  };
  float* bufA = (float*)alloc(64UL * 64 * 64 * 64 * 4);   // h1 / upsampled g
  float* bufB = (float*)alloc(64UL * 128 * 32 * 32 * 4);
  float* bufC = (float*)alloc(64UL * 128 * 32 * 32 * 4);
  float* bufD = (float*)alloc(64UL * 64 * 32 * 32 * 4);   // res-mid / Ze
  float* bufE = (float*)alloc(64UL * 64 * 32 * 32 * 4);   // Zq
  _Float16* Ef16 = (_Float16*)alloc(512UL * 64 * 2);
  float* Enorm = (float*)alloc(512UL * 4);
  int*   idx   = (int*)alloc(65536UL * 4);
  float* counts = (float*)alloc(513UL * 4);  // 512 counts + 1 loss accumulator
  float* accum  = counts + 512;
  // f16 weight copies (B operands for the TDM-staged GEMMs)
  _Float16* wf_e1   = (_Float16*)alloc(64UL * 48 * 2);
  _Float16* wf_e2   = (_Float16*)alloc(128UL * 1024 * 2);
  _Float16* wf_e3   = (_Float16*)alloc(128UL * 1152 * 2);
  _Float16* wf_e4   = (_Float16*)alloc(128UL * 1152 * 2);
  _Float16* wf_er1  = (_Float16*)alloc(2UL * 64 * 1152 * 2);
  _Float16* wf_er2  = (_Float16*)alloc(2UL * 128 * 64 * 2);
  _Float16* wf_eadj = (_Float16*)alloc(64UL * 128 * 2);
  _Float16* wf_dadj = (_Float16*)alloc(128UL * 576 * 2);
  _Float16* wf_dr1  = (_Float16*)alloc(2UL * 64 * 1152 * 2);
  _Float16* wf_dr2  = (_Float16*)alloc(2UL * 128 * 64 * 2);
  _Float16* wf_t1   = (_Float16*)alloc(64UL * 2048 * 2);
  _Float16* wf_t2   = (_Float16*)alloc(3UL * 1024 * 2);

  auto cvt = [&](const float* s, _Float16* d, int n) {
    f32_to_f16_kernel<<<(n + 255) / 256, 256, 0, stream>>>(s, d, n);
  };

  // init counts + accumulator; weight prep
  zero_kernel<<<3, 256, 0, stream>>>(counts, 513);
  cvt(enc_w1, wf_e1, 64 * 48);
  cvt(enc_w2, wf_e2, 128 * 1024);
  cvt(enc_w3, wf_e3, 128 * 1152);
  cvt(enc_w4, wf_e4, 128 * 1152);
  cvt(enc_res_w1, wf_er1, 2 * 64 * 1152);
  cvt(enc_res_w2, wf_er2, 2 * 128 * 64);
  cvt(enc_adj_w, wf_eadj, 64 * 128);
  cvt(dec_adj_w, wf_dadj, 128 * 576);
  cvt(dec_res_w1, wf_dr1, 2 * 64 * 1152);
  cvt(dec_res_w2, wf_dr2, 2 * 128 * 64);
  cvt(E, Ef16, Kc * Dd);
  wtrans_kernel<<<(128 * 64 * 16 + 255) / 256, 256, 0, stream>>>(tc1_w, wf_t1, 128, 64, 4);
  wtrans_kernel<<<(64 * 3 * 16 + 255) / 256, 256, 0, stream>>>(tc2_w, wf_t2, 64, 3, 4);
  enorm_kernel<<<2, 256, 0, stream>>>(E, Enorm, Kc, Dd);

  // ---- Encoder ----
  conv<4, 4, 2, 1, 1>(stream, x,    wf_e1, enc_b1, nullptr, bufA, Bn,   3, 128, 128,  64, 64, 64, 0, 1);
  conv<4, 4, 2, 1, 1>(stream, bufA, wf_e2, enc_b2, nullptr, bufB, Bn,  64,  64,  64, 128, 32, 32, 0, 1);
  conv<3, 3, 1, 1, 1>(stream, bufB, wf_e3, enc_b3, nullptr, bufC, Bn, 128,  32,  32, 128, 32, 32, 0, 1);
  conv<3, 3, 1, 1, 1>(stream, bufC, wf_e4, enc_b4, nullptr, bufB, Bn, 128,  32,  32, 128, 32, 32, 0, 0);
  // res stack (2 blocks)
  const size_t rw1 = 64UL * 1152, rw2 = 128UL * 64;
  conv<3, 3, 1, 1, 1>(stream, bufB, wf_er1,       nullptr, nullptr, bufD, Bn, 128, 32, 32,  64, 32, 32, 1, 0);
  conv<1, 1, 1, 0, 1>(stream, bufD, wf_er2,       nullptr, bufB,    bufC, Bn,  64, 32, 32, 128, 32, 32, 1, 0);
  conv<3, 3, 1, 1, 1>(stream, bufC, wf_er1 + rw1, nullptr, nullptr, bufD, Bn, 128, 32, 32,  64, 32, 32, 1, 0);
  conv<1, 1, 1, 0, 1>(stream, bufD, wf_er2 + rw2, nullptr, bufC,    bufB, Bn,  64, 32, 32, 128, 32, 32, 1, 1);
  // Ze
  conv<1, 1, 1, 0, 1>(stream, bufB, wf_eadj, enc_adj_b, nullptr, bufD, Bn, 128, 32, 32, 64, 32, 32, 0, 0);

  // ---- Vector quantizer ----
  vq_argmin_kernel<<<65536 / 16, 64, 0, stream>>>(bufD, Ef16, Enorm, idx, Bn, Dd, 32, 32, Kc);
  vq_gather_loss_kernel<<<2048, 256, 0, stream>>>(bufD, E, idx, bufE, accum,
                                                  Bn * Dd * 32 * 32, Dd, 32, 32);
  hist_kernel<<<65536 / 256, 256, 0, stream>>>(idx, counts, 65536);

  // ---- Decoder ----
  conv<3, 3, 1, 1, 1>(stream, bufE, wf_dadj, dec_adj_b, nullptr, bufB, Bn, 64, 32, 32, 128, 32, 32, 0, 0);
  conv<3, 3, 1, 1, 1>(stream, bufB, wf_dr1,       nullptr, nullptr, bufD, Bn, 128, 32, 32,  64, 32, 32, 1, 0);
  conv<1, 1, 1, 0, 1>(stream, bufD, wf_dr2,       nullptr, bufB,    bufC, Bn,  64, 32, 32, 128, 32, 32, 1, 0);
  conv<3, 3, 1, 1, 1>(stream, bufC, wf_dr1 + rw1, nullptr, nullptr, bufD, Bn, 128, 32, 32,  64, 32, 32, 1, 0);
  conv<1, 1, 1, 0, 1>(stream, bufD, wf_dr2 + rw2, nullptr, bufC,    bufB, Bn,  64, 32, 32, 128, 32, 32, 1, 1);
  // transposed convs: pre-flipped weights, lhs_dilation=2, pad=k-1-p=2, stride 1
  conv<4, 4, 1, 2, 2>(stream, bufB, wf_t1, tc1_b, nullptr, bufA,    Bn, 128, 32, 32,  64,  64,  64, 0, 1);
  conv<4, 4, 1, 2, 2>(stream, bufA, wf_t2, tc2_b, nullptr, out + 1, Bn,  64, 64, 64,   3, 128, 128, 0, 0);

  // ---- scalars ----
  finalize_kernel<<<1, 512, 0, stream>>>(counts, accum, out, R, Kc, 65536.0f,
                                         65536.0f * 64.0f);
}
